// Cam2BEV_72086731096598
// MI455X (gfx1250) — matrix operand order repaired and checked by reading, not compiled, not run on hardware
//
#include <hip/hip_runtime.h>
#include <hip/hip_bf16.h>

// ---------------- problem constants ----------------
#define Bv       2
#define Nv       6
#define NIMG     12          // B*N
#define FC       256
#define FHc      32
#define FWc      88
#define Pv       2816        // FH*FW
#define Dv       20          // depths 1..58 step 3
#define OUTC     128
#define COUT3    130
#define COUT3P   144         // padded to 9*16
#define KT       8           // 256/32 k-tiles
#define PT       176         // 2816/16 spatial tiles (1x1 conv)
#define WT6      6           // ceil(88/16) w-tiles (3x3 conv)
#define NXv      256
#define NYv      256
#define BEVSZ    (Bv*OUTC*NXv*NYv)   // 16,777,216
#define MASKSZ   (Bv*NXv*NYv)        // 131,072

typedef unsigned short u16;
typedef unsigned int   u32;
typedef __bf16 bf16;
typedef bf16  v16bf __attribute__((ext_vector_type(16)));
typedef float v8f   __attribute__((ext_vector_type(8)));
typedef u32   v8u   __attribute__((ext_vector_type(8)));
typedef int   v4i   __attribute__((vector_size(16)));

#define AS1 __attribute__((address_space(1)))
#define AS3 __attribute__((address_space(3)))

#if defined(__has_builtin)
#  if __has_builtin(__builtin_amdgcn_global_load_async_to_lds_b128)
#    define HAVE_ASYNC_LDS 1
#  endif
#endif

// generic -> address-space pointers via integer casts (value-correct on AMDGPU:
// as1 == full 64-bit VA; as3 == low 32 bits of the generic LDS address)
__device__ __forceinline__ AS1 v4i* to_global_v4i(const void* p) {
    return (AS1 v4i*)(unsigned long long)p;
}
__device__ __forceinline__ AS3 v4i* to_lds_v4i(void* p) {
    return (AS3 v4i*)(unsigned int)(unsigned long long)p;
}

// ---------------- helpers ----------------
__device__ __forceinline__ u16 f2bf(float f) {
    u32 x = __builtin_bit_cast(u32, f);
    u32 r = x + 0x7FFFu + ((x >> 16) & 1u);   // round-to-nearest-even
    return (u16)(r >> 16);
}
__device__ __forceinline__ v16bf load_frag32B(const u16* p) {
    uint4 lo = *(const uint4*)p;
    uint4 hi = *(const uint4*)(p + 8);
    v8u t; t[0]=lo.x; t[1]=lo.y; t[2]=lo.z; t[3]=lo.w;
           t[4]=hi.x; t[5]=hi.y; t[6]=hi.z; t[7]=hi.w;
    return __builtin_bit_cast(v16bf, t);
}

// ---------------- utility kernels ----------------
__global__ void k_zero(float* p, int n) {
    int i = blockIdx.x * blockDim.x + threadIdx.x;
    if (i < n) p[i] = 0.0f;
}

// img_feats (NIMG, FC, FH, FW) f32 -> channels-last bf16 [img][p][ci]
__global__ void k_prep_input(const float* __restrict__ src, u16* __restrict__ dst) {
    int i = blockIdx.x * blockDim.x + threadIdx.x;
    const int total = NIMG * FC * Pv;
    if (i >= total) return;
    int img = i / (FC * Pv);
    int r   = i - img * (FC * Pv);
    int ci  = r / Pv;
    int p   = r - ci * Pv;
    dst[(size_t)(img * Pv + p) * FC + ci] = f2bf(src[i]);
}

// pack conv weights (Cout,Cin,KHW) f32 into WMMA A-fragment layout bf16:
// frag f = (mt*KHW+pos)*KT' + kt ; within frag: lane(32) x elem(16)
// A-layout: m = lane&15 ; K = e + (lane>>4)*16
__global__ void k_pack_w(const float* __restrict__ w, u16* __restrict__ dst,
                         int Cout, int CoutPad, int Cin, int KHW) {
    int i = blockIdx.x * blockDim.x + threadIdx.x;
    int ktiles = Cin / 32;
    int total = (CoutPad / 16) * KHW * ktiles * 512;
    if (i >= total) return;
    int f    = i >> 9;           // /512
    int r    = i & 511;
    int lane = r >> 4;
    int e    = r & 15;
    int kt   = f % ktiles;
    int t2   = f / ktiles;
    int pos  = t2 % KHW;
    int mt   = t2 / KHW;
    int khalf = lane >> 4;
    int co = mt * 16 + (lane & 15);
    int kk = e + khalf * 16;
    int ci = kt * 32 + kk;
    float v = (co < Cout) ? w[((size_t)co * Cin + ci) * KHW + pos] : 0.0f;
    dst[i] = f2bf(v);
}

// fold conv bias into BN affine: a = s/sqrt(v+eps); b = (bias_bn - m*a) + conv_b*a
__global__ void k_prep_bn(const float* s1, const float* b1, const float* m1, const float* v1,
                          const float* cb1,
                          const float* s2, const float* b2, const float* m2, const float* v2,
                          const float* cb2, float* bnbuf) {
    int c = blockIdx.x * blockDim.x + threadIdx.x;
    if (c >= FC) return;
    float a1 = s1[c] * __frsqrt_rn(v1[c] + 1e-3f);
    bnbuf[c]           = a1;
    bnbuf[FC + c]      = (b1[c] - m1[c] * a1) + cb1[c] * a1;
    float a2 = s2[c] * __frsqrt_rn(v2[c] + 1e-3f);
    bnbuf[2 * FC + c]  = a2;
    bnbuf[3 * FC + c]  = (b2[c] - m2[c] * a2) + cb2[c] * a2;
}

// per-camera 3x3 inverses:  matbuf[idx*24] = {invpost[9], combine[9], c2e_t[3], post_t[3]}
__device__ void inv3(const float* m, float* o) {
    float a=m[0],b=m[1],c=m[2],d=m[3],e=m[4],f=m[5],g=m[6],h=m[7],i=m[8];
    float A =  e*i - f*h, Bq = -(d*i - f*g), C =  d*h - e*g;
    float det = a*A + b*Bq + c*C;
    float r = 1.0f / det;
    o[0]=A*r;            o[1]=-(b*i-c*h)*r;  o[2]=(b*f-c*e)*r;
    o[3]=Bq*r;           o[4]=(a*i-c*g)*r;   o[5]=-(a*f-c*d)*r;
    o[6]=C*r;            o[7]=-(a*h-b*g)*r;  o[8]=(a*e-b*d)*r;
}
__global__ void k_mats(const float* c2e_rot, const float* c2e_t, const float* intrins,
                       const float* post_rot, const float* post_t, float* matbuf) {
    int idx = threadIdx.x;
    if (idx >= NIMG) return;
    float* o = matbuf + idx * 24;
    inv3(post_rot + idx * 9, o);              // invpost -> [0..8]
    float Ki[9];
    inv3(intrins + idx * 9, Ki);
    const float* R = c2e_rot + idx * 9;       // combine = R @ Ki -> [9..17]
    for (int r = 0; r < 3; ++r)
        for (int c = 0; c < 3; ++c)
            o[9 + r * 3 + c] = R[r*3+0]*Ki[0*3+c] + R[r*3+1]*Ki[1*3+c] + R[r*3+2]*Ki[2*3+c];
    o[18] = c2e_t[idx*3+0]; o[19] = c2e_t[idx*3+1]; o[20] = c2e_t[idx*3+2];
    o[21] = post_t[idx*3+0]; o[22] = post_t[idx*3+1]; o[23] = post_t[idx*3+2];
}

// ---------------- WMMA 3x3 conv + BN + ReLU (bf16 in/out, f32 acc) ----------------
// Cooperative workgroup version:
//   grid = (WT6, FHc, NIMG), block = 512 threads (16 waves).
//   Input tile (3 rows x 18 cols x 256ch, zero-padded halo) is staged to LDS once
//   via GLOBAL_LOAD_ASYNC_TO_LDS_B128 (ASYNCcnt) and shared by all 16 waves;
//   wave w computes output-channel tile mt=w. Inner loop is branch-free:
//   2x ds_load_b128 (B) + 2x global_load_b128 (A) + v_wmma per step.
#define TILE_RC   (3 * 18)          // 54 staged (row,col) positions
#define TILE_U16  (TILE_RC * FC)    // 13,824 u16 = 27,648 B of LDS

__global__ void __launch_bounds__(512)
k_conv3x3_wmma(const u16* __restrict__ xin, const u16* __restrict__ wpk,
               const float* __restrict__ bnA, const float* __restrict__ bnB,
               u16* __restrict__ xout) {
    __shared__ u16 tile[TILE_U16];

    const int tid   = threadIdx.x;
    const int lane  = tid & 31;
    const int mt    = tid >> 5;          // wave id == output-channel tile
    const int khalf = lane >> 4;
    const int nlo   = lane & 15;
    const int wt    = blockIdx.x;        // w-tile (16 wide, last one partial)
    const int h     = blockIdx.y;        // output row
    const int img   = blockIdx.z;

    // ---- stage input halo tile into LDS (zero-fill out-of-bounds) ----
    for (int q = tid; q < TILE_RC * 32; q += 512) {     // 16B chunks
        int rc  = q >> 5;
        int ciq = q & 31;
        int r   = rc / 18, c = rc - r * 18;
        int hh  = h - 1 + r;
        int ww  = wt * 16 - 1 + c;
        u16* ldst = &tile[rc * FC + ciq * 8];
        bool ok = (hh >= 0) & (hh < FHc) & (ww >= 0) & (ww < FWc);
        if (ok) {
            const u16* g = xin + ((size_t)(img * Pv + hh * FWc + ww) * FC + ciq * 8);
#ifdef HAVE_ASYNC_LDS
            __builtin_amdgcn_global_load_async_to_lds_b128(
                to_global_v4i(g), to_lds_v4i(ldst), 0, 0);
#else
            *(uint4*)ldst = *(const uint4*)g;
#endif
        } else {
            uint4 z = {0u, 0u, 0u, 0u};
            *(uint4*)ldst = z;
        }
    }
#ifdef HAVE_ASYNC_LDS
#  if __has_builtin(__builtin_amdgcn_s_wait_asynccnt)
    __builtin_amdgcn_s_wait_asynccnt(0);
#  else
    asm volatile("s_wait_asynccnt 0" ::: "memory");
#  endif
#endif
    __syncthreads();

    // ---- 72-step WMMA K-loop, B from LDS, A from global (L2-resident) ----
    v8f acc = {};
    const u16* wbase = wpk + (size_t)(mt * 9) * (KT * 512);
    for (int pos = 0; pos < 9; ++pos) {
        const int kh = pos / 3, kw = pos - kh * 3;
        const u16* bbase  = &tile[((kh * 18) + nlo + kw) * FC + khalf * 16];
        const u16* afrag0 = wbase + (size_t)pos * (KT * 512);
        for (int kt = 0; kt < KT; ++kt) {
            __builtin_prefetch(afrag0 + (kt + 1) * 512 + lane * 16, 0, 0);
            v16bf a = load_frag32B(afrag0 + kt * 512 + lane * 16);
            v16bf b = load_frag32B(bbase + kt * 32);
            acc = __builtin_amdgcn_wmma_f32_16x16x32_bf16(
                      false, a, false, b, (short)0, acc, false, false);
        }
    }

    // ---- fused BN + ReLU, pack 8 consecutive channels, one 16B store ----
    const int wcol = wt * 16 + nlo;
    if (wcol >= FWc) return;             // padded columns of last w-tile
    const int p = h * FWc + wcol;
    const int cobase = mt * 16 + khalf * 8;
    u16 hv[8];
    #pragma unroll
    for (int r = 0; r < 8; ++r) {
        int co = cobase + r;
        float y = acc[r] * bnA[co] + bnB[co];
        y = y > 0.0f ? y : 0.0f;
        hv[r] = f2bf(y);
    }
    uint4 st;
    st.x = (u32)hv[0] | ((u32)hv[1] << 16);
    st.y = (u32)hv[2] | ((u32)hv[3] << 16);
    st.z = (u32)hv[4] | ((u32)hv[5] << 16);
    st.w = (u32)hv[6] | ((u32)hv[7] << 16);
    *(uint4*)(xout + (size_t)(img * Pv + p) * FC + cobase) = st;
}

// ---------------- WMMA 1x1 conv (256 -> 130, padded 144) ----------------
// grid = (PT, COUT3P/16, NIMG), block = 32
__global__ void __launch_bounds__(32)
k_conv1x1_wmma(const u16* __restrict__ xin, const u16* __restrict__ wpk,
               const float* __restrict__ b3,
               float* __restrict__ mu, float* __restrict__ logs,
               float* __restrict__ feats) {
    const int lane  = threadIdx.x;
    const int khalf = lane >> 4;
    const int nlo   = lane & 15;
    const int ptile = blockIdx.x;
    const int mt    = blockIdx.y;
    const int img   = blockIdx.z;
    const int p     = ptile * 16 + nlo;

    v8f acc = {};
    const u16* bbase  = xin + ((size_t)(img * Pv + p) * FC + khalf * 16);
    const u16* afrag0 = wpk + (size_t)mt * (KT * 512);
    for (int kt = 0; kt < KT; ++kt) {
        v16bf a = load_frag32B(afrag0 + kt * 512 + lane * 16);
        v16bf b = load_frag32B(bbase + kt * 32);
        acc = __builtin_amdgcn_wmma_f32_16x16x32_bf16(
                  false, a, false, b, (short)0, acc, false, false);
    }

    const int cobase = mt * 16 + khalf * 8;
    const int pp = img * Pv + p;
    #pragma unroll
    for (int r = 0; r < 8; ++r) {
        int co = cobase + r;
        if (co >= COUT3) continue;
        float y = acc[r] + b3[co];
        if (co == 0)      mu[pp]   = y;
        else if (co == 1) logs[pp] = y;
        else              feats[(size_t)pp * OUTC + (co - 2)] = y;
    }
}

// ---------------- depth gaussian softmax: g[img][p][d] ----------------
__global__ void k_depth(const float* __restrict__ mu, const float* __restrict__ logs,
                        float* __restrict__ g) {
    int i = blockIdx.x * blockDim.x + threadIdx.x;
    if (i >= NIMG * Pv) return;
    float m = mu[i];
    float sg = __expf(logs[i]) + 1e-6f;
    float inv2 = -0.5f / (sg * sg);
    float e[Dv], sum = 0.0f;
    #pragma unroll
    for (int d = 0; d < Dv; ++d) {
        float dv = 1.0f + 3.0f * (float)d;
        float t = dv - m;
        e[d] = __expf(inv2 * t * t);
        sum += e[d];
    }
    float r = 1.0f / (sum + 1e-6f);
    #pragma unroll
    for (int d = 0; d < Dv; ++d) g[(size_t)i * Dv + d] = e[d] * r;
}

// ---------------- frustum -> voxel flat index (or -1) ----------------
__global__ void k_frustum(const float* __restrict__ matbuf, int* __restrict__ flatidx) {
    int i = blockIdx.x * blockDim.x + threadIdx.x;
    const int total = NIMG * Dv * Pv;
    if (i >= total) return;
    int img = i / (Dv * Pv);
    int r   = i - img * (Dv * Pv);
    int d   = r / Pv;
    int p   = r - d * Pv;
    int h = p / FWc, w = p - (p / FWc) * FWc;
    const float* M = matbuf + img * 24;

    float fx = (float)w * (703.0f / 87.0f);
    float fy = (float)h * (255.0f / 31.0f);
    float fd = 1.0f + 3.0f * (float)d;

    float px = fx - M[21], py = fy - M[22], pz = fd - M[23];
    float qx = M[0]*px + M[1]*py + M[2]*pz;
    float qy = M[3]*px + M[4]*py + M[5]*pz;
    float qz = M[6]*px + M[7]*py + M[8]*pz;
    qx *= qz; qy *= qz;
    float gx = M[ 9]*qx + M[10]*qy + M[11]*qz + M[18];
    float gy = M[12]*qx + M[13]*qy + M[14]*qz + M[19];
    float gz = M[15]*qx + M[16]*qy + M[17]*qz + M[20];

    int ix = (int)((gx + 51.2f) * 2.5f);   // (g - (bx - dx/2)) / dx , dx=0.4 bx=-51
    int iy = (int)((gy + 51.2f) * 2.5f);
    int iz = (int)((gz + 10.0f) * 0.05f);  // dx.z=20, bx.z=0
    bool kept = (ix >= 0) & (ix < NXv) & (iy >= 0) & (iy < NYv) & (iz == 0);
    int b = img / Nv;
    flatidx[i] = kept ? (b * (NXv * NYv) + ix * NYv + iy) : -1;
}

// ---------------- scatter: bev[b][c][gx][gy] += g * feat ----------------
// grid = NIMG*Dv*Pv blocks, block = 128 (one channel per lane)
__global__ void __launch_bounds__(128)
k_scatter(const int* __restrict__ flatidx, const float* __restrict__ g,
          const float* __restrict__ feats, float* __restrict__ bev,
          float* __restrict__ counts) {
    int i = blockIdx.x;
    int flat = flatidx[i];
    if (flat < 0) return;
    int img = i / (Dv * Pv);
    int r   = i - img * (Dv * Pv);
    int d   = r / Pv;
    int p   = r - d * Pv;
    int pp  = img * Pv + p;
    float gv = g[(size_t)pp * Dv + d];
    int c = threadIdx.x;
    float val = gv * feats[(size_t)pp * OUTC + c];
    int b  = flat >> 16;            // NX*NY == 65536
    int xy = flat & 0xFFFF;
    atomicAdd(bev + ((size_t)b * OUTC + c) * (NXv * NYv) + xy, val);
    if (c == 0) atomicAdd(counts + flat, 1.0f);
}

__global__ void k_mask(const float* __restrict__ counts, float* __restrict__ maskout) {
    int i = blockIdx.x * blockDim.x + threadIdx.x;
    if (i >= MASKSZ) return;
    maskout[i] = counts[i] > 0.0f ? 1.0f : 0.0f;
}

// ---------------- host launcher ----------------
extern "C" void kernel_launch(void* const* d_in, const int* in_sizes, int n_in,
                              void* d_out, int out_size, void* d_ws, size_t ws_size,
                              hipStream_t stream) {
    const float* c2e_rot  = (const float*)d_in[0];
    const float* c2e_t    = (const float*)d_in[1];
    const float* intrins  = (const float*)d_in[2];
    const float* post_rot = (const float*)d_in[3];
    const float* post_t   = (const float*)d_in[4];
    const float* imgf     = (const float*)d_in[5];
    const float* w1       = (const float*)d_in[6];
    const float* cb1      = (const float*)d_in[7];
    const float* bn1s     = (const float*)d_in[8];
    const float* bn1b     = (const float*)d_in[9];
    const float* bn1m     = (const float*)d_in[10];
    const float* bn1v     = (const float*)d_in[11];
    const float* w2       = (const float*)d_in[12];
    const float* cb2      = (const float*)d_in[13];
    const float* bn2s     = (const float*)d_in[14];
    const float* bn2b     = (const float*)d_in[15];
    const float* bn2m     = (const float*)d_in[16];
    const float* bn2v     = (const float*)d_in[17];
    const float* w3       = (const float*)d_in[18];
    const float* cb3      = (const float*)d_in[19];
    float* out = (float*)d_out;

    // workspace carve-up (256B aligned)
    char* ws = (char*)d_ws;
    size_t off = 0;
    auto carve = [&](size_t bytes) -> char* {
        char* p = ws + off;
        off = (off + bytes + 255) & ~(size_t)255;
        return p;
    };
    const size_t SZX = (size_t)NIMG * Pv * FC * 2;          // bf16 activations
    u16*   bufA   = (u16*)  carve(SZX);
    u16*   bufB   = (u16*)  carve(SZX);
    u16*   w1pk   = (u16*)  carve((size_t)16 * 9 * KT * 512 * 2);
    u16*   w2pk   = (u16*)  carve((size_t)16 * 9 * KT * 512 * 2);
    u16*   w3pk   = (u16*)  carve((size_t)9 * 1 * KT * 512 * 2);
    float* mu     = (float*)carve((size_t)NIMG * Pv * 4);
    float* logs   = (float*)carve((size_t)NIMG * Pv * 4);
    float* feats  = (float*)carve((size_t)NIMG * Pv * OUTC * 4);
    float* gbuf   = (float*)carve((size_t)NIMG * Pv * Dv * 4);
    int*   flati  = (int*)  carve((size_t)NIMG * Dv * Pv * 4);
    float* counts = (float*)carve((size_t)MASKSZ * 4);
    float* bnbuf  = (float*)carve((size_t)4 * FC * 4);
    float* matbuf = (float*)carve((size_t)NIMG * 24 * 4);
    (void)ws_size; (void)in_sizes; (void)n_in; (void)out_size;

    // 1) zero output + counts
    k_zero<<<(BEVSZ + MASKSZ + 255) / 256, 256, 0, stream>>>(out, BEVSZ + MASKSZ);
    k_zero<<<(MASKSZ + 255) / 256, 256, 0, stream>>>(counts, MASKSZ);

    // 2) input & weight repack to WMMA fragment layouts
    {
        int tot = NIMG * FC * Pv;
        k_prep_input<<<(tot + 255) / 256, 256, 0, stream>>>(imgf, bufA);
    }
    {
        int tot = 16 * 9 * KT * 512;
        k_pack_w<<<(tot + 255) / 256, 256, 0, stream>>>(w1, w1pk, FC, FC, FC, 9);
        k_pack_w<<<(tot + 255) / 256, 256, 0, stream>>>(w2, w2pk, FC, FC, FC, 9);
    }
    {
        int tot = 9 * 1 * KT * 512;
        k_pack_w<<<(tot + 255) / 256, 256, 0, stream>>>(w3, w3pk, COUT3, COUT3P, FC, 1);
    }
    k_prep_bn<<<1, FC, 0, stream>>>(bn1s, bn1b, bn1m, bn1v, cb1,
                                    bn2s, bn2b, bn2m, bn2v, cb2, bnbuf);
    k_mats<<<1, 32, 0, stream>>>(c2e_rot, c2e_t, intrins, post_rot, post_t, matbuf);

    // 3) conv1 -> conv2 -> conv3 (WMMA; 3x3 convs use LDS-staged cooperative tiles)
    dim3 cg(WT6, FHc, NIMG);
    k_conv3x3_wmma<<<cg, 512, 0, stream>>>(bufA, w1pk, bnbuf,          bnbuf + FC,     bufB);
    k_conv3x3_wmma<<<cg, 512, 0, stream>>>(bufB, w2pk, bnbuf + 2 * FC, bnbuf + 3 * FC, bufA);
    dim3 cg3(PT, COUT3P / 16, NIMG);
    k_conv1x1_wmma<<<cg3, 32, 0, stream>>>(bufA, w3pk, cb3, mu, logs, feats);

    // 4) depth softmax, geometry, scatter, mask
    {
        int tot = NIMG * Pv;
        k_depth<<<(tot + 255) / 256, 256, 0, stream>>>(mu, logs, gbuf);
    }
    {
        int tot = NIMG * Dv * Pv;
        k_frustum<<<(tot + 255) / 256, 256, 0, stream>>>(matbuf, flati);
        k_scatter<<<tot, 128, 0, stream>>>(flati, gbuf, feats, out, counts);
    }
    k_mask<<<(MASKSZ + 255) / 256, 256, 0, stream>>>(counts, out + BEVSZ);
}
